// BiNet_70506183131681
// MI455X (gfx1250) — matrix operand order repaired and checked by reading, not compiled, not run on hardware
//
#include <hip/hip_runtime.h>
#include <hip/hip_bf16.h>
#include <math.h>

typedef __attribute__((ext_vector_type(16))) _Float16 v16h;
typedef __attribute__((ext_vector_type(8)))  _Float16 v8h;
typedef __attribute__((ext_vector_type(8)))  float    v8f;
typedef __attribute__((ext_vector_type(4)))  float    v4f;

#define C_CH   32
#define K_IN   68
#define K_PAD  96
#define WPB    4      // waves per block (wave32)
#define TPW    8      // consecutive tiles per wave (amortizes weight staging)
// branch lut: mod 0..10 -> {0,1,2,2,2,2,3,3,3,1,0}, 2 bits each
#define BR_LUT 522916u
__device__ __forceinline__ int branch_of(int dst) {
    return (int)((BR_LUT >> (2u * ((unsigned)dst % 11u))) & 3u);
}

// ---------------- binning kernels ----------------

__global__ void bin_count_kernel(const int* __restrict__ edge_dst, int E, int Ecap,
                                 int* __restrict__ counts, int* __restrict__ reorder) {
    int i = blockIdx.x * blockDim.x + threadIdx.x;
    if (i < Ecap) reorder[i] = -1;           // sentinel fill (padding slots)
    if (i < E) atomicAdd(&counts[branch_of(edge_dst[i])], 1);
}

__global__ void bin_offsets_kernel(const int* __restrict__ counts, int* __restrict__ cursors) {
    if (threadIdx.x == 0 && blockIdx.x == 0) {
        int off = 0;
        for (int b = 0; b < 4; ++b) {
            cursors[b] = off;
            off += (counts[b] + 15) & ~15;   // 16-align each branch region
        }
    }
}

__global__ void bin_scatter_kernel(const int* __restrict__ edge_dst, int E,
                                   int* __restrict__ cursors, int* __restrict__ reorder) {
    int i = blockIdx.x * blockDim.x + threadIdx.x;
    if (i >= E) return;
    int pos = atomicAdd(&cursors[branch_of(edge_dst[i])], 1);
    reorder[pos] = i;
}

// ---------------- edge-tile WMMA MLP kernel ----------------

__device__ __forceinline__ float angle3(float ax, float ay, float az,
                                        float bx, float by, float bz) {
    float cx = ay * bz - az * by;
    float cy = az * bx - ax * bz;
    float cz = ax * by - ay * bx;
    float cn = sqrtf(cx * cx + cy * cy + cz * cz);
    float dt = ax * bx + ay * by + az * bz;
    return atan2f(cn, dt);
}

__global__ void __launch_bounds__(WPB * 32)
edge_mlp_kernel(const float* __restrict__ x_dna,  const float* __restrict__ v_dna,
                const float* __restrict__ x_prot, const float* __restrict__ v_prot,
                const float* __restrict__ prot_vec, const float* __restrict__ dna_vec,
                const int* __restrict__ edge_src, const int* __restrict__ edge_dst,
                const float* __restrict__ W1, const float* __restrict__ b1,
                const float* __restrict__ W2, const float* __restrict__ b2,
                const int* __restrict__ reorder, int numTiles,
                float* __restrict__ conv) {
    // per-wave private LDS slices
    __shared__ _Float16 sA [WPB][16 * K_PAD]   __attribute__((aligned(64))); // feat tile, K padded
    __shared__ _Float16 sW1[WPB][C_CH * K_PAD] __attribute__((aligned(64))); // W1^T [n][k]
    __shared__ _Float16 sW2[WPB][C_CH * C_CH]  __attribute__((aligned(64))); // W2^T [n][k]
    __shared__ _Float16 sH [WPB][16 * C_CH]    __attribute__((aligned(64))); // hidden tile
    __shared__ int sId [WPB][16];
    __shared__ int sSrc[WPB][16];
    __shared__ int sDst[WPB][16];

    const int lane = threadIdx.x & 31;
    const int wave = threadIdx.x >> 5;
    const int m    = lane & 15;
    const int aBase = (lane >= 16) ? 16 : 0;   // A-frag half-row select
    const int kbOff = (lane >= 16) ? 16 : 0;   // B-frag K-half select
    const int mRow  = (lane >= 16) ? 8 : 0;    // C/D-frag M-half select

    const int tile0 = (blockIdx.x * WPB + wave) * TPW;
    if (tile0 >= numTiles) return;             // wave-uniform
    const int tileEnd = (tile0 + TPW < numTiles) ? tile0 + TPW : numTiles;

    int brCur = -1;                            // cached staged branch
    float bias1a = 0.f, bias1b = 0.f, bias2a = 0.f, bias2b = 0.f;

    for (int tile = tile0; tile < tileEnd; ++tile) {
        // ---- phase 1: fetch the 16 edge ids of this tile ----
        if (lane < 16) {
            int e = reorder[tile * 16 + lane];
            int s = 0, d = -1;
            if (e >= 0) { s = edge_src[e]; d = edge_dst[e]; }
            sId[wave][lane] = e; sSrc[wave][lane] = s; sDst[wave][lane] = d;
        }
        asm volatile("s_wait_dscnt 0" ::: "memory");

        int firstDst = -1;
        for (int j = 0; j < 16 && firstDst < 0; ++j) firstDst = sDst[wave][j];
        if (firstDst < 0) break;               // fully-padded tile => past end of edges
        const int br = branch_of(firstDst);

        // ---- stage branch weights (transposed, f16) only when branch changes ----
        if (br != brCur) {
            brCur = br;
            for (int t = lane; t < C_CH * K_PAD; t += 32) {
                int n = t / K_PAD, k = t % K_PAD;
                float w = (k < K_IN) ? W1[((size_t)br * K_IN + k) * C_CH + n] : 0.f;
                sW1[wave][n * K_PAD + k] = (_Float16)w;
            }
            for (int t = lane; t < C_CH * C_CH; t += 32) {
                int n = t >> 5, k = t & 31;
                sW2[wave][n * C_CH + k] = (_Float16)W2[((size_t)br * C_CH + k) * C_CH + n];
            }
            bias1a = b1[br * C_CH + m];
            bias1b = b1[br * C_CH + 16 + m];
            bias2a = b2[br * C_CH + m];
            bias2b = b2[br * C_CH + 16 + m];
        }

        // ---- phase 2: build A tile [16 x 96] f16 (feat = [x_prot | ppf | x_dna]) ----
        const int eId = sId[wave][m];
        const int src = sSrc[wave][m];
        const int dst = sDst[wave][m];
        if (lane < 16) {
            if (eId >= 0) {
                float px = v_prot[src * 3 + 0], py = v_prot[src * 3 + 1], pz = v_prot[src * 3 + 2];
                float qx = v_dna [dst * 3 + 0], qy = v_dna [dst * 3 + 1], qz = v_dna [dst * 3 + 2];
                float dx = px - qx, dy = py - qy, dz = pz - qz;          // pos_j - pos_i
                float nix = dna_vec [dst * 3 + 0], niy = dna_vec [dst * 3 + 1], niz = dna_vec [dst * 3 + 2];
                float njx = prot_vec[src * 3 + 0], njy = prot_vec[src * 3 + 1], njz = prot_vec[src * 3 + 2];
                float p0 = sqrtf(dx * dx + dy * dy + dz * dz);
                float p1 = angle3(nix, niy, niz, dx, dy, dz);
                float p2 = angle3(njx, njy, njz, dx, dy, dz);
                float p3 = angle3(nix, niy, niz, njx, njy, njz);
                #pragma unroll
                for (int k = 0; k < C_CH; k += 4) {
                    v4f x = *(const v4f*)&x_prot[(size_t)src * C_CH + k];
                    sA[wave][m * K_PAD + k + 0] = (_Float16)x[0];
                    sA[wave][m * K_PAD + k + 1] = (_Float16)x[1];
                    sA[wave][m * K_PAD + k + 2] = (_Float16)x[2];
                    sA[wave][m * K_PAD + k + 3] = (_Float16)x[3];
                }
                sA[wave][m * K_PAD + 32] = (_Float16)p0;
                sA[wave][m * K_PAD + 33] = (_Float16)p1;
                sA[wave][m * K_PAD + 34] = (_Float16)p2;
                sA[wave][m * K_PAD + 35] = (_Float16)p3;
            } else {
                for (int k = 0; k < 36; ++k) sA[wave][m * K_PAD + k] = (_Float16)0.f;
            }
        } else {
            if (eId >= 0) {
                #pragma unroll
                for (int k = 0; k < C_CH; k += 4) {
                    v4f x = *(const v4f*)&x_dna[(size_t)dst * C_CH + k];
                    sA[wave][m * K_PAD + 36 + k + 0] = (_Float16)x[0];
                    sA[wave][m * K_PAD + 36 + k + 1] = (_Float16)x[1];
                    sA[wave][m * K_PAD + 36 + k + 2] = (_Float16)x[2];
                    sA[wave][m * K_PAD + 36 + k + 3] = (_Float16)x[3];
                }
            } else {
                for (int k = 0; k < C_CH; ++k) sA[wave][m * K_PAD + 36 + k] = (_Float16)0.f;
            }
            for (int k = K_IN; k < K_PAD; ++k) sA[wave][m * K_PAD + k] = (_Float16)0.f;
        }
        asm volatile("s_wait_dscnt 0" ::: "memory");

        // ---- MLP1: acc[16x32] = relu(A[16x96] * W1[96x32] + b1) via 6 WMMAs ----
        v8f acc0, acc1;
        #pragma unroll
        for (int r = 0; r < 8; ++r) { acc0[r] = bias1a; acc1[r] = bias1b; }
        #pragma unroll
        for (int k0 = 0; k0 < K_PAD; k0 += 32) {
            v8h lo = *(const v8h*)&sA[wave][m * K_PAD + k0 + aBase];
            v8h hi = *(const v8h*)&sA[wave][m * K_PAD + k0 + aBase + 8];
            v16h a;
            #pragma unroll
            for (int i = 0; i < 8; ++i) { a[i] = lo[i]; a[i + 8] = hi[i]; }
            v16h bA = *(const v16h*)&sW1[wave][m * K_PAD + k0 + kbOff];
            v16h bB = *(const v16h*)&sW1[wave][(16 + m) * K_PAD + k0 + kbOff];
            acc0 = __builtin_amdgcn_wmma_f32_16x16x32_f16(false, a, false, bA, (short)0, acc0, false, false);
            acc1 = __builtin_amdgcn_wmma_f32_16x16x32_f16(false, a, false, bB, (short)0, acc1, false, false);
        }

        // ---- relu, re-stage hidden tile as f16 A-operand ----
        #pragma unroll
        for (int r = 0; r < 8; ++r) {
            float t0 = acc0[r] > 0.f ? acc0[r] : 0.f;
            float t1 = acc1[r] > 0.f ? acc1[r] : 0.f;
            sH[wave][(mRow + r) * C_CH + m]      = (_Float16)t0;
            sH[wave][(mRow + r) * C_CH + 16 + m] = (_Float16)t1;
        }
        asm volatile("s_wait_dscnt 0" ::: "memory");

        // ---- MLP2: out[16x32] = relu(H[16x32] * W2[32x32] + b2) via 2 WMMAs ----
        v8f o0, o1;
        #pragma unroll
        for (int r = 0; r < 8; ++r) { o0[r] = bias2a; o1[r] = bias2b; }
        {
            v8h lo = *(const v8h*)&sH[wave][m * C_CH + aBase];
            v8h hi = *(const v8h*)&sH[wave][m * C_CH + aBase + 8];
            v16h a;
            #pragma unroll
            for (int i = 0; i < 8; ++i) { a[i] = lo[i]; a[i + 8] = hi[i]; }
            v16h bA = *(const v16h*)&sW2[wave][m * C_CH + kbOff];
            v16h bB = *(const v16h*)&sW2[wave][(16 + m) * C_CH + kbOff];
            o0 = __builtin_amdgcn_wmma_f32_16x16x32_f16(false, a, false, bA, (short)0, o0, false, false);
            o1 = __builtin_amdgcn_wmma_f32_16x16x32_f16(false, a, false, bB, (short)0, o1, false, false);
        }

        // ---- segment-sum via f32 global atomics (skip padding rows) ----
        #pragma unroll
        for (int r = 0; r < 8; ++r) {
            int mm = mRow + r;
            int d = sDst[wave][mm];
            if (d >= 0) {
                float t0 = o0[r] > 0.f ? o0[r] : 0.f;
                float t1 = o1[r] > 0.f ? o1[r] : 0.f;
                atomicAdd(&conv[(size_t)d * C_CH + m],      t0);
                atomicAdd(&conv[(size_t)d * C_CH + 16 + m], t1);
            }
        }
    }
}

// ---------------- finalize: out = relu(conv) ----------------

__global__ void finalize_kernel(const float* __restrict__ conv, float* __restrict__ out, int n) {
    int i = blockIdx.x * blockDim.x + threadIdx.x;
    if (i < n) { float v = conv[i]; out[i] = v > 0.f ? v : 0.f; }
}

// ---------------- launch ----------------

extern "C" void kernel_launch(void* const* d_in, const int* in_sizes, int n_in,
                              void* d_out, int out_size, void* d_ws, size_t ws_size,
                              hipStream_t stream) {
    const float* x_dna    = (const float*)d_in[0];
    const float* v_dna    = (const float*)d_in[1];
    const float* x_prot   = (const float*)d_in[2];
    const float* v_prot   = (const float*)d_in[3];
    const float* prot_vec = (const float*)d_in[4];
    const float* dna_vec  = (const float*)d_in[5];
    const int*   edge_src = (const int*)d_in[6];
    const int*   edge_dst = (const int*)d_in[7];
    const float* W1       = (const float*)d_in[8];
    const float* b1       = (const float*)d_in[9];
    const float* W2       = (const float*)d_in[10];
    const float* b2       = (const float*)d_in[11];

    const int E     = in_sizes[6];
    const int N_DNA = in_sizes[0] / C_CH;
    const int nOut  = N_DNA * C_CH;

    float* out  = (float*)d_out;
    float* conv = out + (size_t)nOut;      // second tuple element accumulates here

    int* counts  = (int*)d_ws;             // [0..3] counts, [4..7] cursors
    int* cursors = counts + 4;
    int* reorder = (int*)((char*)d_ws + 256);
    const int Ecap = ((E + 60) + 15) & ~15;   // room for 16-aligned branch regions

    hipMemsetAsync(counts, 0, 32, stream);
    hipMemsetAsync(conv, 0, (size_t)nOut * sizeof(float), stream);

    bin_count_kernel<<<(Ecap + 255) / 256, 256, 0, stream>>>(edge_dst, E, Ecap, counts, reorder);
    bin_offsets_kernel<<<1, 64, 0, stream>>>(counts, cursors);
    bin_scatter_kernel<<<(E + 255) / 256, 256, 0, stream>>>(edge_dst, E, cursors, reorder);

    const int numTiles = Ecap / 16;
    const int wavesNeeded = (numTiles + TPW - 1) / TPW;
    edge_mlp_kernel<<<(wavesNeeded + WPB - 1) / WPB, WPB * 32, 0, stream>>>(
        x_dna, v_dna, x_prot, v_prot, prot_vec, dna_vec,
        edge_src, edge_dst, W1, b1, W2, b2, reorder, numTiles, conv);

    finalize_kernel<<<(nOut + 255) / 256, 256, 0, stream>>>(conv, out, nOut);
}
